// HOI_Detector_17145509445800
// MI455X (gfx1250) — compile-verified, matrix operand order checked by reading, not run on hardware
//
#include <hip/hip_runtime.h>
#include <stdint.h>

// ---- problem constants -----------------------------------------------------
#define BB   16
#define PP   12
#define OO   17
#define DD   1024
#define NPAIR 3264            // BB*PP*OO
#define KPAIR 3104            // 3076 padded up to a multiple of 32
#define KGRAPH 2048

typedef __attribute__((ext_vector_type(16))) __bf16 v16bf;
typedef __attribute__((ext_vector_type(8)))  __bf16 v8bf;
typedef __attribute__((ext_vector_type(8)))  float  v8f;

__device__ __forceinline__ __bf16 f2bf(float f) {
  union { float f; uint32_t u; } v; v.f = f;
  uint32_t r = v.u + 0x7FFFu + ((v.u >> 16) & 1u);   // round-to-nearest-even
  uint16_t h = (uint16_t)(r >> 16);
  __bf16 b; __builtin_memcpy(&b, &h, 2); return b;
}

// Load one 16-element bf16 fragment slice: two 16-byte chunks at +0 and +16
// (matches CDNA5 16-bit A/B layout: lane<16 -> K{0..7,16..23}, lane>=16 -> K{8..15,24..31})
__device__ __forceinline__ v16bf ldfrag(const __bf16* p) {
  v8bf lo = *(const v8bf*)(p);
  v8bf hi = *(const v8bf*)(p + 16);
  return __builtin_shufflevector(lo, hi, 0,1,2,3,4,5,6,7,8,9,10,11,12,13,14,15);
}

// ---- generic WMMA GEMM:  out = act((A(MxK) * W(NxK)^T)*scale + bias) [+res] -
// A, W row-major bf16; K % 32 == 0, M % 16 == 0, N % 64 == 0.
// Each wave computes a 16x64 tile; 8 waves / block.
__global__ __launch_bounds__(256)
void gemm_bf16_wmma(const __bf16* __restrict__ A, const __bf16* __restrict__ W,
                    void* __restrict__ out, int M, int N, int K,
                    const float* __restrict__ scale,   // BN gamma (x rsqrt(1+eps)) or null
                    const float* __restrict__ bias,    // beta / linear bias or null
                    const float* __restrict__ resid,   // f32 NCHW residual or null
                    int hw, int relu, int out_bf16)
{
  const float RS = rsqrtf(1.0f + 1e-5f);
  int tid  = threadIdx.x;
  int lane = tid & 31;
  int wid  = tid >> 5;
  int gw   = blockIdx.x * 8 + wid;
  int tilesN = N >> 6;
  int mt = gw / tilesN;
  int nt = gw - mt * tilesN;
  if (mt * 16 >= M) return;

  int r  = mt * 16 + (lane & 15);
  int ks = (lane >> 4) * 8;
  const __bf16* Ap = A + (size_t)r * K + ks;
  int cb = nt * 64 + (lane & 15);
  const __bf16* W0 = W + (size_t)(cb +  0) * K + ks;
  const __bf16* W1 = W + (size_t)(cb + 16) * K + ks;
  const __bf16* W2 = W + (size_t)(cb + 32) * K + ks;
  const __bf16* W3 = W + (size_t)(cb + 48) * K + ks;

  v8f c0 = {}, c1 = {}, c2 = {}, c3 = {};
  for (int k0 = 0; k0 < K; k0 += 32) {
    v16bf a  = ldfrag(Ap + k0);
    v16bf b0 = ldfrag(W0 + k0);
    v16bf b1 = ldfrag(W1 + k0);
    v16bf b2 = ldfrag(W2 + k0);
    v16bf b3 = ldfrag(W3 + k0);
    c0 = __builtin_amdgcn_wmma_f32_16x16x32_bf16(false, a, false, b0, (short)0, c0, false, false);
    c1 = __builtin_amdgcn_wmma_f32_16x16x32_bf16(false, a, false, b1, (short)0, c1, false, false);
    c2 = __builtin_amdgcn_wmma_f32_16x16x32_bf16(false, a, false, b2, (short)0, c2, false, false);
    c3 = __builtin_amdgcn_wmma_f32_16x16x32_bf16(false, a, false, b3, (short)0, c3, false, false);
  }

  int rowBase = mt * 16 + ((lane >> 4) * 8);
  for (int j = 0; j < 4; ++j) {
    v8f c = (j == 0) ? c0 : (j == 1) ? c1 : (j == 2) ? c2 : c3;
    int col = nt * 64 + j * 16 + (lane & 15);
    float sc = 1.0f, bi = 0.0f;
    if (scale) sc = scale[col] * RS;
    if (bias)  bi = bias[col];
    for (int e = 0; e < 8; ++e) {
      int row = rowBase + e;
      float v = c[e] * sc + bi;
      if (relu) v = v > 0.0f ? v : 0.0f;
      if (resid) {
        int n = row / hw, p = row - n * hw;
        v += resid[(size_t)n * N * hw + (size_t)col * hw + p];
      }
      if (out_bf16) ((__bf16*)out)[(size_t)row * N + col] = f2bf(v);
      else          ((float*) out)[(size_t)row * N + col] = v;
    }
  }
}

// ---- conversion kernels ----------------------------------------------------
__global__ void cvt_pad_bf16(const float* __restrict__ s, __bf16* __restrict__ d,
                             int N, int Kin, int Kout)
{
  size_t idx = (size_t)blockIdx.x * 256 + threadIdx.x;
  size_t tot = (size_t)N * Kout;
  if (idx >= tot) return;
  int n = (int)(idx / Kout), k = (int)(idx - (size_t)n * Kout);
  d[idx] = f2bf(k < Kin ? s[(size_t)n * Kin + k] : 0.0f);
}

__global__ void cvt_nchw_bf16(const float* __restrict__ x, __bf16* __restrict__ xt,
                              int NI, int C, int HW)
{
  size_t idx = (size_t)blockIdx.x * 256 + threadIdx.x;
  size_t tot = (size_t)NI * C * HW;
  if (idx >= tot) return;
  int p = (int)(idx % HW);
  size_t t = idx / HW;
  int c = (int)(t % C);
  int n = (int)(t / C);
  xt[((size_t)n * HW + p) * C + c] = f2bf(x[idx]);
}

__global__ void cvt_objno0_bf16(const float* __restrict__ obj, __bf16* __restrict__ d)
{
  int idx = blockIdx.x * 256 + threadIdx.x;        // 256*1024
  if (idx >= 256 * 1024) return;
  int row = idx >> 10, k = idx & 1023;
  int b = row >> 4, o = row & 15;
  d[idx] = f2bf(obj[((size_t)(b * 17 + o + 1)) * 1024 + k]);
}

// ---- mean over HW of row-major [N*HW, 1024] --------------------------------
__global__ void meanpool_k(const float* __restrict__ in, float* __restrict__ out,
                           int NI, int HW)
{
  int idx = blockIdx.x * 256 + threadIdx.x;
  if (idx >= NI * 1024) return;
  int n = idx >> 10, c = idx & 1023;
  float acc = 0.0f;
  const float* p = in + ((size_t)n * HW) * 1024 + c;
  for (int q = 0; q < HW; ++q) acc += p[(size_t)q * 1024];
  out[idx] = acc / (float)HW;
}

// ---- fused spatial stream: conv1+pool+conv2+pool+mean+up-proj per image ----
__global__ __launch_bounds__(256)
void spatial_fused(const float* __restrict__ ub,
                   const float* __restrict__ c1w, const float* __restrict__ c1b,
                   const float* __restrict__ c2w, const float* __restrict__ c2b,
                   const float* __restrict__ upw, const float* __restrict__ upb,
                   float* __restrict__ u2)
{
  extern __shared__ float lds[];
  float* img = lds;                 // 2*64*64 = 8192
  float* w1s = lds + 8192;          // 3200
  float* b1s = w1s + 3200;          // 64
  float* sp  = b1s + 64;            // 32
  float* c1p = sp + 32;             // 64*30*30 = 57600
  int tid = threadIdx.x;
  int i = blockIdx.x;
  const float* src = ub + (size_t)i * 8192;
  for (int t = tid; t < 8192; t += 256) img[t] = src[t];
  for (int t = tid; t < 3200; t += 256) w1s[t] = c1w[t];
  for (int t = tid; t < 64;   t += 256) b1s[t] = c1b[t];
  if (tid < 32) sp[tid] = 0.0f;
  __syncthreads();

  // conv1 (2->64, 5x5 valid: 64->60) fused with 2x2 maxpool -> (64,30,30)
  for (int t = tid; t < 64 * 30 * 30; t += 256) {
    int oc = t / 900;
    int rem = t - oc * 900;
    int oy = rem / 30, ox = rem - (rem / 30) * 30;
    float m = -3.4e38f;
    for (int dy = 0; dy < 2; ++dy)
      for (int dx = 0; dx < 2; ++dx) {
        int y = oy * 2 + dy, x = ox * 2 + dx;
        float acc = b1s[oc];
        for (int ic = 0; ic < 2; ++ic) {
          const float* ipx = img + ic * 4096 + y * 64 + x;
          const float* wp  = w1s + oc * 50 + ic * 25;
          for (int ky = 0; ky < 5; ++ky)
            for (int kx = 0; kx < 5; ++kx)
              acc += ipx[ky * 64 + kx] * wp[ky * 5 + kx];
        }
        m = fmaxf(m, acc);
      }
    c1p[t] = m;
  }
  __syncthreads();

  // conv2 (64->32, 5x5: 30->26) fused with pool -> (32,13,13) and mean over 169
  for (int t = tid; t < 32 * 13 * 13; t += 256) {
    int oc = t / 169;
    int rem = t - oc * 169;
    int oy = rem / 13, ox = rem - (rem / 13) * 13;
    float m = -3.4e38f;
    for (int dy = 0; dy < 2; ++dy)
      for (int dx = 0; dx < 2; ++dx) {
        int y = oy * 2 + dy, x = ox * 2 + dx;
        float acc = c2b[oc];
        for (int ic = 0; ic < 64; ++ic) {
          const float* ipx = c1p + ic * 900 + y * 30 + x;
          const float* wp  = c2w + ((size_t)(oc * 64 + ic)) * 25;
          for (int ky = 0; ky < 5; ++ky)
            for (int kx = 0; kx < 5; ++kx)
              acc += ipx[ky * 30 + kx] * wp[ky * 5 + kx];
        }
        m = fmaxf(m, acc);
      }
    atomicAdd(&sp[oc], m * (1.0f / 169.0f));
  }
  __syncthreads();

  // out2_union = relu(sp @ up_w^T + up_b)
  for (int j = tid; j < 512; j += 256) {
    float acc = upb[j];
    for (int k = 0; k < 32; ++k) acc += sp[k] * upw[j * 32 + k];
    u2[(size_t)i * 512 + j] = fmaxf(acc, 0.0f);
  }
}

// ---- pair tensors ----------------------------------------------------------
__global__ void build_pairs(const float* __restrict__ peo, const float* __restrict__ obj,
                            const float* __restrict__ ctx, const float* __restrict__ sl,
                            __bf16* __restrict__ pairs)
{
  size_t idx = (size_t)blockIdx.x * 256 + threadIdx.x;
  size_t tot = (size_t)NPAIR * KPAIR;
  if (idx >= tot) return;
  int r = (int)(idx / KPAIR), c = (int)(idx - (size_t)r * KPAIR);
  int b = r / 204, rem = r - b * 204, p = rem / 17, o = rem - (rem / 17) * 17;
  float v;
  if      (c < 1024) v = peo[((size_t)(b * 12 + p)) * 1024 + c];
  else if (c < 2048) v = obj[((size_t)(b * 17 + o)) * 1024 + (c - 1024)];
  else if (c < 3072) v = ctx[(size_t)b * 1024 + (c - 2048)];
  else if (c < 3076) v = sl[(size_t)r * 4 + (c - 3072)];
  else               v = 0.0f;
  pairs[idx] = f2bf(v);
}

__global__ void build_pairs_graph(const float* __restrict__ people_r,
                                  const float* __restrict__ obj_full,
                                  __bf16* __restrict__ pg)
{
  size_t idx = (size_t)blockIdx.x * 256 + threadIdx.x;
  size_t tot = (size_t)NPAIR * KGRAPH;
  if (idx >= tot) return;
  int r = (int)(idx / KGRAPH), c = (int)(idx - (size_t)r * KGRAPH);
  int b = r / 204, rem = r - b * 204, p = rem / 17, o = rem - (rem / 17) * 17;
  float v = (c < 1024) ? people_r[((size_t)(b * 12 + p)) * 1024 + c]
                       : obj_full[((size_t)(b * 17 + o)) * 1024 + (c - 1024)];
  pg[idx] = f2bf(v);
}

// ---- small output heads: out[r,j] = sum_k (S?S*U:U)[r,k] * Wt[j,k] + bt[j] -
__global__ __launch_bounds__(256)
void head_small(const float* __restrict__ S, const float* __restrict__ U,
                const float* __restrict__ Wt, const float* __restrict__ bt,
                float* __restrict__ out, int n_out, float* __restrict__ ip)
{
  __shared__ float s[512];
  int r = blockIdx.x, tid = threadIdx.x;
  for (int k = tid; k < 512; k += 256) {
    float u = U[(size_t)r * 512 + k];
    s[k] = S ? S[(size_t)r * 512 + k] * u : u;
  }
  __syncthreads();
  for (int j = tid; j < n_out; j += 256) {
    float acc = bt[j];
    const float* w = Wt + (size_t)j * 512;
    for (int k = 0; k < 512; ++k) acc += s[k] * w[k];
    out[(size_t)r * n_out + j] = acc;
    if (ip) ip[r] = 1.0f / (1.0f + __expf(-acc));
  }
}

// ---- graph message aggregation ---------------------------------------------
__global__ void agg_people(const float* __restrict__ peo, const float* __restrict__ ipb,
                           const float* __restrict__ msg_o, float* __restrict__ people_r)
{
  int idx = blockIdx.x * 256 + threadIdx.x;       // 192*1024
  if (idx >= 192 * 1024) return;
  int row = idx >> 10, k = idx & 1023;
  int b = row / 12, p = row - b * 12;
  float acc = peo[idx];
  for (int o = 0; o < 16; ++o)
    acc += ipb[(size_t)(b * 12 + p) * 17 + o + 1] * msg_o[((size_t)(b * 16 + o)) * 1024 + k];
  people_r[idx] = acc;
}

__global__ void agg_obj(const float* __restrict__ obj, const float* __restrict__ ipb,
                        const float* __restrict__ msg_p, float* __restrict__ obj_full)
{
  int idx = blockIdx.x * 256 + threadIdx.x;       // 272*1024
  if (idx >= 272 * 1024) return;
  int row = idx >> 10, k = idx & 1023;
  int b = row / 17, o = row - b * 17;
  float acc = obj[idx];
  if (o > 0) {
    for (int p = 0; p < 12; ++p) {
      int fi = (o - 1) * 12 + p;                  // adj_op = adj_po.reshape(B,16,12)
      float a = ipb[(size_t)(b * 12 + fi / 16) * 17 + (fi % 16) + 1];
      acc += a * msg_p[((size_t)(b * 12 + p)) * 1024 + k];
    }
  }
  obj_full[idx] = acc;
}

// ---- host launcher ---------------------------------------------------------
extern "C" void kernel_launch(void* const* d_in, const int* in_sizes, int n_in,
                              void* d_out, int out_size, void* d_ws, size_t ws_size,
                              hipStream_t stream)
{
  (void)in_sizes; (void)n_in; (void)out_size; (void)ws_size;
#define IN(i) ((const float*)d_in[(i)])
  const float* rois_people  = IN(0);
  const float* rois_objects = IN(1);
  const float* out1         = IN(2);
  const float* union_box    = IN(3);
  const float* spatial_locs = IN(4);
  // params (setup_inputs insertion order)
  // people:5..13  objects:14..22  context:23..31
  const float* sp_c1_w = IN(32); const float* sp_c1_b = IN(33);
  const float* sp_c2_w = IN(34); const float* sp_c2_b = IN(35);
  const float* up_w = IN(36);    const float* up_b = IN(37);
  const float* att_w = IN(38);   const float* att_b = IN(39);
  const float* po_w = IN(40);    const float* po_b = IN(41);
  const float* op_w = IN(42);    const float* op_b = IN(43);
  const float* sg_w1 = IN(44);   const float* sg_b1 = IN(45);
  const float* sg_w2 = IN(46);   const float* sg_b2 = IN(47);
  const float* sg_wt = IN(48);   const float* sg_bt = IN(49);
  const float* vs_w1 = IN(50);   const float* vs_b1 = IN(51);
  const float* vs_w2 = IN(52);   const float* vs_b2 = IN(53);
  const float* vs_wt = IN(54);   const float* vs_bt = IN(55);
  const float* gr_w1 = IN(56);   const float* gr_b1 = IN(57);
  const float* gr_w2 = IN(58);   const float* gr_b2 = IN(59);
  const float* gr_wt = IN(60);   const float* gr_bt = IN(61);

  char* ws = (char*)d_ws;
  size_t off = 0;
  auto alloc = [&](size_t bytes) -> void* {
    void* p = ws + off;
    off = (off + bytes + 255) & ~(size_t)255;
    return p;
  };

  // bf16 weight buffers
  __bf16* brw1[3]; __bf16* brw2[3]; __bf16* brw3[3];
  for (int i = 0; i < 3; ++i) {
    brw1[i] = (__bf16*)alloc((size_t)512 * 1024 * 2);
    brw2[i] = (__bf16*)alloc((size_t)512 * 512 * 2);
    brw3[i] = (__bf16*)alloc((size_t)1024 * 512 * 2);
  }
  __bf16* po_wb  = (__bf16*)alloc((size_t)1024 * 1024 * 2);
  __bf16* op_wb  = (__bf16*)alloc((size_t)1024 * 1024 * 2);
  __bf16* sg_w1b = (__bf16*)alloc((size_t)1024 * KPAIR * 2);
  __bf16* vs_w1b = (__bf16*)alloc((size_t)1024 * KPAIR * 2);
  __bf16* gr_w1b = (__bf16*)alloc((size_t)1024 * KGRAPH * 2);
  __bf16* sg_w2b = (__bf16*)alloc((size_t)512 * 1024 * 2);
  __bf16* vs_w2b = (__bf16*)alloc((size_t)512 * 1024 * 2);
  __bf16* gr_w2b = (__bf16*)alloc((size_t)512 * 1024 * 2);

  // activations
  __bf16* xt_p = (__bf16*)alloc((size_t)19200 * 1024 * 2);
  __bf16* xt_o = (__bf16*)alloc((size_t)27200 * 1024 * 2);
  __bf16* xt_c = (__bf16*)alloc((size_t)10000 * 1024 * 2);
  __bf16* h1   = (__bf16*)alloc((size_t)27200 * 512 * 2);
  __bf16* h2   = (__bf16*)alloc((size_t)27200 * 512 * 2);
  float*  bo   = (float*) alloc((size_t)27200 * 1024 * 4);
  float*  o2p  = (float*) alloc((size_t)192 * 1024 * 4);
  float*  o2o  = (float*) alloc((size_t)272 * 1024 * 4);
  float*  o2c  = (float*) alloc((size_t)16 * 1024 * 4);
  float*  u2   = (float*) alloc((size_t)NPAIR * 512 * 4);
  __bf16* pairs = (__bf16*)alloc((size_t)NPAIR * KPAIR * 2);
  __bf16* tmp1  = (__bf16*)alloc((size_t)NPAIR * 1024 * 2);
  float*  sgb  = (float*) alloc((size_t)NPAIR * 512 * 4);
  float*  vsb  = (float*) alloc((size_t)NPAIR * 512 * 4);
  float*  grb  = (float*) alloc((size_t)NPAIR * 512 * 4);
  float*  ipb  = (float*) alloc((size_t)NPAIR * 4);
  __bf16* objno0b = (__bf16*)alloc((size_t)256 * 1024 * 2);
  __bf16* peob    = (__bf16*)alloc((size_t)192 * 1024 * 2);
  float*  msg_o   = (float*) alloc((size_t)256 * 1024 * 4);
  float*  msg_p   = (float*) alloc((size_t)192 * 1024 * 4);
  float*  people_r = (float*)alloc((size_t)192 * 1024 * 4);
  float*  obj_full = (float*)alloc((size_t)272 * 1024 * 4);
  __bf16* pg = (__bf16*)alloc((size_t)NPAIR * KGRAPH * 2);

  auto cvtw = [&](const float* s, __bf16* d, int N, int Kin, int Kout) {
    size_t tot = (size_t)N * Kout;
    cvt_pad_bf16<<<(unsigned)((tot + 255) / 256), 256, 0, stream>>>(s, d, N, Kin, Kout);
  };
  auto gemm = [&](const __bf16* A, const __bf16* W, void* out, int M, int N, int K,
                  const float* scale, const float* bias, const float* resid, int hw,
                  int relu, int obf) {
    int tiles = (M / 16) * (N / 64);
    gemm_bf16_wmma<<<(tiles + 7) / 8, 256, 0, stream>>>(A, W, out, M, N, K,
                                                        scale, bias, resid, hw, relu, obf);
  };

  // --- weight conversions ---
  for (int i = 0; i < 3; ++i) {
    const float* base5 = IN(5 + 9 * i);        // w1 of this branch
    cvtw(base5, brw1[i], 512, 1024, 1024);
    cvtw(IN(5 + 9 * i + 3), brw2[i], 512, 512, 512);
    cvtw(IN(5 + 9 * i + 6), brw3[i], 1024, 512, 512);
  }
  cvtw(po_w, po_wb, 1024, 1024, 1024);
  cvtw(op_w, op_wb, 1024, 1024, 1024);
  cvtw(sg_w1, sg_w1b, 1024, 3076, KPAIR);
  cvtw(vs_w1, vs_w1b, 1024, 3076, KPAIR);
  cvtw(gr_w1, gr_w1b, 1024, 2048, KGRAPH);
  cvtw(sg_w2, sg_w2b, 512, 1024, 1024);
  cvtw(vs_w2, vs_w2b, 512, 1024, 1024);
  cvtw(gr_w2, gr_w2b, 512, 1024, 1024);

  // --- input transposes (NCHW -> [N*HW, C] bf16) ---
  {
    size_t t;
    t = (size_t)192 * 1024 * 100;
    cvt_nchw_bf16<<<(unsigned)((t + 255) / 256), 256, 0, stream>>>(rois_people, xt_p, 192, 1024, 100);
    t = (size_t)272 * 1024 * 100;
    cvt_nchw_bf16<<<(unsigned)((t + 255) / 256), 256, 0, stream>>>(rois_objects, xt_o, 272, 1024, 100);
    t = (size_t)16 * 1024 * 625;
    cvt_nchw_bf16<<<(unsigned)((t + 255) / 256), 256, 0, stream>>>(out1, xt_c, 16, 1024, 625);
  }

  // --- residual branches (GEMM + BN fused, final layer fuses relu+residual) ---
  auto run_branch = [&](const float* x, const __bf16* xt, int rows, int hw,
                        int pb /*param base*/, __bf16* w1b, __bf16* w2b, __bf16* w3b,
                        float* out2) {
    gemm(xt, w1b, h1, rows, 512, 1024, IN(pb + 1), IN(pb + 2), nullptr, 0, 0, 1);
    gemm(h1, w2b, h2, rows, 512, 512,  IN(pb + 4), IN(pb + 5), nullptr, 0, 0, 1);
    gemm(h2, w3b, bo, rows, 1024, 512, IN(pb + 7), IN(pb + 8), x, hw, 1, 0);
    int ni = rows / hw;
    meanpool_k<<<(ni * 1024 + 255) / 256, 256, 0, stream>>>(bo, out2, ni, hw);
  };
  run_branch(rois_people,  xt_p, 19200, 100, 5,  brw1[0], brw2[0], brw3[0], o2p);
  run_branch(rois_objects, xt_o, 27200, 100, 14, brw1[1], brw2[1], brw3[1], o2o);
  run_branch(out1,         xt_c, 10000, 625, 23, brw1[2], brw2[2], brw3[2], o2c);

  // --- spatial stream (fused per image; 276,352 B dynamic LDS) ---
  spatial_fused<<<NPAIR, 256, 276352, stream>>>(union_box, sp_c1_w, sp_c1_b,
                                                sp_c2_w, sp_c2_b, up_w, up_b, u2);

  // --- pairs and sg head ---
  {
    size_t t = (size_t)NPAIR * KPAIR;
    build_pairs<<<(unsigned)((t + 255) / 256), 256, 0, stream>>>(o2p, o2o, o2c, spatial_locs, pairs);
  }
  gemm(pairs, sg_w1b, tmp1, NPAIR, 1024, KPAIR, nullptr, sg_b1, nullptr, 0, 0, 1);
  gemm(tmp1, sg_w2b, sgb, NPAIR, 512, 1024, nullptr, sg_b2, nullptr, 0, 1, 0);

  float* d_lv = (float*)d_out;            // lin_visual (3264 x 29)
  float* d_ls = d_lv + (size_t)NPAIR * 29; // lin_single (3264 x 1)
  float* d_lg = d_ls + (size_t)NPAIR;      // lin_graph  (3264 x 29)
  float* d_la = d_lg + (size_t)NPAIR * 29; // lin_att    (3264 x 29)

  head_small<<<NPAIR, 256, 0, stream>>>(sgb, u2, sg_wt, sg_bt, d_ls, 1, ipb);

  // --- message passing ---
  cvt_objno0_bf16<<<(256 * 1024) / 256, 256, 0, stream>>>(o2o, objno0b);
  gemm(objno0b, po_wb, msg_o, 256, 1024, 1024, nullptr, po_b, nullptr, 0, 1, 0);
  cvtw(o2p, peob, 192, 1024, 1024);
  gemm(peob, op_wb, msg_p, 192, 1024, 1024, nullptr, op_b, nullptr, 0, 1, 0);
  agg_people<<<(192 * 1024) / 256, 256, 0, stream>>>(o2p, ipb, msg_o, people_r);
  agg_obj<<<(272 * 1024) / 256, 256, 0, stream>>>(o2o, ipb, msg_p, obj_full);
  {
    size_t t = (size_t)NPAIR * KGRAPH;
    build_pairs_graph<<<(unsigned)((t + 255) / 256), 256, 0, stream>>>(people_r, obj_full, pg);
  }

  // --- vs head ---
  gemm(pairs, vs_w1b, tmp1, NPAIR, 1024, KPAIR, nullptr, vs_b1, nullptr, 0, 0, 1);
  gemm(tmp1, vs_w2b, vsb, NPAIR, 512, 1024, nullptr, vs_b2, nullptr, 0, 1, 0);
  head_small<<<NPAIR, 256, 0, stream>>>(vsb, u2, vs_wt, vs_bt, d_lv, 29, nullptr);

  // --- gr head ---
  gemm(pg, gr_w1b, tmp1, NPAIR, 1024, KGRAPH, nullptr, gr_b1, nullptr, 0, 0, 1);
  gemm(tmp1, gr_w2b, grb, NPAIR, 512, 1024, nullptr, gr_b2, nullptr, 0, 1, 0);
  head_small<<<NPAIR, 256, 0, stream>>>(grb, u2, gr_wt, gr_bt, d_lg, 29, nullptr);

  // --- att head ---
  head_small<<<NPAIR, 256, 0, stream>>>(nullptr, u2, att_w, att_b, d_la, 29, nullptr);
#undef IN
}
